// Net_view_47665547051341
// MI455X (gfx1250) — matrix operand order repaired and checked by reading, not compiled, not run on hardware
//
#include <hip/hip_runtime.h>

typedef __attribute__((ext_vector_type(16))) _Float16 v16h;
typedef __attribute__((ext_vector_type(8)))  _Float16 v8h;
typedef __attribute__((ext_vector_type(8)))  float    v8f;

__host__ __device__ constexpr int rup(int x, int m) { return ((x + m - 1) / m) * m; }

// ------------------------------------------------------------------
// Weight prep: f32 [OC][IC][KS][KS] -> f16 [OCpad][Kpad], zero padded.
// K index = ic*KS*KS + kh*KS + kw  (same order the im2col gather uses).
// ------------------------------------------------------------------
template<int IC, int OC, int KS>
__global__ void prep_w_kernel(const float* __restrict__ w, _Float16* __restrict__ w16) {
  constexpr int K   = IC * KS * KS;
  constexpr int Kp  = rup(K, 64);
  constexpr int OCp = rup(OC, 16);
  int i = blockIdx.x * 256 + threadIdx.x;
  if (i >= OCp * Kp) return;
  int oc = i / Kp, k = i % Kp;
  float v = (oc < OC && k < K) ? w[oc * K + k] : 0.0f;
  w16[i] = (_Float16)v;
}

// ------------------------------------------------------------------
// Zero-padded bilinear sample on a 64x64 f32 image (matches reference
// semantics: valid iff coord in [0, 63], taps clipped).
// ------------------------------------------------------------------
__device__ inline float bilin64(const float* __restrict__ img, float x, float y) {
  float x0 = floorf(x), y0 = floorf(y);
  float wx = x - x0, wy = y - y0;
  float s = 0.0f;
#pragma unroll
  for (int j = 0; j < 2; ++j) {
#pragma unroll
    for (int i = 0; i < 2; ++i) {
      float xf = x0 + (float)i, yf = y0 + (float)j;
      if (xf >= 0.0f && xf <= 63.0f && yf >= 0.0f && yf <= 63.0f) {
        float wgt = (i ? wx : 1.0f - wx) * (j ? wy : 1.0f - wy);
        s += wgt * img[(int)yf * 64 + (int)xf];
      }
    }
  }
  return s;
}

// ------------------------------------------------------------------
// Plane-sweep volume, already cropped to 56x56, stored f16 as
// [tb*16 images][4 src channels][3136 pixels]  (image = tl*16 + d).
// ------------------------------------------------------------------
__global__ void build_psv_kernel(const float* __restrict__ img, const int* __restrict__ ind,
                                 _Float16* __restrict__ psv, int t0, int tb) {
  int idx = blockIdx.x * 256 + threadIdx.x;
  if (idx >= tb * 16 * 4 * 3136) return;
  int pix = idx % 3136; int r = idx / 3136;
  int s = r % 4; r /= 4;
  int d = r % 16; int tl = r / 16;
  int t = t0 + tl;
  float disp = -4.0f + (8.0f / 15.0f) * (float)d;   // linspace(-4, 4, 16)
  int iv = ind[s];
  float dy = (float)(iv / 7) - (float)(t / 7);
  float dx = (float)(iv % 7) - (float)(t % 7);
  int y = pix / 56, x = pix % 56;
  float v = bilin64(img + s * 4096, (float)(x + 4) + disp * dx, (float)(y + 4) + disp * dy);
  psv[idx] = (_Float16)v;
}

// ------------------------------------------------------------------
// Implicit-GEMM conv via v_wmma_f32_16x16x32_f16, double-buffered LDS.
//   in  : f16 [B][IC][3136]      out : f16 [B][OC][3136]
//   w16 : f16 [OCpad][Kpad]      bias: f32 [OC]
// Per workgroup (8 waves): 128-pixel N macro-tile. Per K-chunk:
//   * A tile (OCp x KC weights) copied global->LDS with
//     GLOBAL_LOAD_ASYNC_TO_LDS_B128 (ASYNCcnt), shared by all 8 waves.
//   * B tile (128 pixels x KC im2col) gathered cooperatively into LDS,
//     K-contiguous per pixel so B fragments are contiguous ds_load_b128.
// Both tiles are double-buffered: the fill for chunk c+1 is issued before
// the WMMA work on chunk c; s_wait_asynccnt 0 + barrier close each stage.
// ------------------------------------------------------------------
template<int IC, int OC, int KS, bool RELU>
__global__ __launch_bounds__(256) void conv_wmma_kernel(
    const _Float16* __restrict__ in, const _Float16* __restrict__ w16,
    const float* __restrict__ bias, _Float16* __restrict__ out, int B) {
  constexpr int HW = 56, NPIX = HW * HW;
  constexpr int K   = IC * KS * KS;
  constexpr int Kp  = rup(K, 64);
  constexpr int OCp = rup(OC, 16);
  constexpr int MT  = OCp / 16;
  constexpr int PAD = KS / 2;
  constexpr int NT  = 128;                      // pixels per workgroup
  constexpr int KC  = (OCp > 128) ? 32 : 64;    // K chunk; keeps LDS <= 48 KB
  constexpr int NC  = Kp / KC;                  // Kp is a multiple of 64
  constexpr int SEG = KC / 8;                   // 16-byte transfers per A row

  __shared__ __attribute__((aligned(128))) _Float16 Bs[2][NT][KC];
  __shared__ __attribute__((aligned(128))) _Float16 As[2][OCp][KC];

  const int b    = blockIdx.y;
  const int p0   = blockIdx.x * NT;
  const int tid  = threadIdx.x;
  const int wave = tid >> 5;
  const int lane = tid & 31;
  const int ln16 = lane & 15;
  const bool hi  = lane >= 16;

  const _Float16* inB = in + (size_t)b * IC * NPIX;

  v8f acc[MT];
#pragma unroll
  for (int m = 0; m < MT; ++m) { v8f z = {0,0,0,0,0,0,0,0}; acc[m] = z; }

  // Stage the A (weights) and B (im2col) tiles for K range [kb, kb+KC).
  auto fill = [&](int kb, int buf) {
    // A: contiguous per-row copies, async DMA into LDS (no VGPR round-trip).
    {
      uint32_t ldsBase = (uint32_t)(uintptr_t)&As[buf][0][0];
      for (int u = tid; u < OCp * SEG; u += 256) {
        int r = u / SEG, seg = u % SEG;
        uint32_t la = ldsBase + (uint32_t)(r * KC + seg * 8) * 2u;
        unsigned long long ga =
            (unsigned long long)(uintptr_t)(w16 + (size_t)r * Kp + kb + seg * 8);
        asm volatile("global_load_async_to_lds_b128 %0, %1, off"
                     :: "v"(la), "v"(ga) : "memory");
      }
    }
    // B: boundary-aware im2col gather (zero padding), NT*KC f16 elements.
    for (int i = tid; i < NT * KC; i += 256) {
      int pl = i / KC, kk = i % KC;
      int gk = kb + kk, gp = p0 + pl;
      _Float16 v = (_Float16)0.0f;
      if (gk < K && gp < NPIX) {
        int y  = gp / HW, x = gp % HW;
        int ic = gk / (KS * KS);
        int r  = gk % (KS * KS);
        int iy = y + r / KS - PAD;
        int ix = x + r % KS - PAD;
        if (iy >= 0 && iy < HW && ix >= 0 && ix < HW)
          v = inB[(size_t)ic * NPIX + iy * HW + ix];
      }
      Bs[buf][pl][kk] = v;
    }
  };

  fill(0, 0);
  asm volatile("s_wait_asynccnt 0x0" ::: "memory");
  __syncthreads();

  for (int c = 0; c < NC; ++c) {
    const int buf = c & 1;
    if (c + 1 < NC) fill((c + 1) * KC, buf ^ 1);   // overlaps with WMMA below

#pragma unroll
    for (int ks = 0; ks < KC; ks += 32) {
      // B 32x16 fragment: lane = column n=lane%16, K = ks + (hi?16:0) + 0..15
      const _Float16* bp = &Bs[buf][wave * 16 + ln16][ks + (hi ? 16 : 0)];
      v8h blo = *(const v8h*)bp;
      v8h bhi = *(const v8h*)(bp + 8);
      v16h bfrag;
#pragma unroll
      for (int e = 0; e < 8; ++e) { bfrag[e] = blo[e]; bfrag[e + 8] = bhi[e]; }

#pragma unroll
      for (int m = 0; m < MT; ++m) {
        // A 16x32 fragment: row m*16+lane%16; halves 0..15 = K base+0..15,
        // base = hi ? 16 : 0.
        const _Float16* ap = &As[buf][m * 16 + ln16][ks + (hi ? 16 : 0)];
        v8h alo = *(const v8h*)ap;
        v8h ahi = *(const v8h*)(ap + 8);
        v16h afrag;
#pragma unroll
        for (int e = 0; e < 8; ++e) { afrag[e] = alo[e]; afrag[e + 8] = ahi[e]; }
        acc[m] = __builtin_amdgcn_wmma_f32_16x16x32_f16(
            /*neg_a=*/false, afrag, /*neg_b=*/false, bfrag,
            /*c_mod=*/(short)0, acc[m], /*reuse_a=*/false, /*reuse_b=*/false);
      }
    }
    asm volatile("s_wait_asynccnt 0x0" ::: "memory");  // next A tile landed
    __syncthreads();                                   // next B tile + buffer reuse
  }

  // C/D layout: reg r, lane L -> (m = r + 8*(L>=16), n = L%16)
  const int pn = p0 + wave * 16 + ln16;
  if (pn < NPIX) {
#pragma unroll
    for (int m = 0; m < MT; ++m) {
#pragma unroll
      for (int r = 0; r < 8; ++r) {
        int oc = m * 16 + r + (hi ? 8 : 0);
        if (oc < OC) {
          float v = acc[m][r] + bias[oc];
          if (RELU) v = fmaxf(v, 0.0f);
          out[(size_t)b * OC * NPIX + (size_t)oc * NPIX + pn] = (_Float16)v;
        }
      }
    }
  }
}

// ------------------------------------------------------------------
// Heads: disparity (padded to 64x64) and softmax confidence.
// dp layout: f16 [tb][5][3136]; channel 0 = disparity, 1..4 = logits.
// ------------------------------------------------------------------
__global__ void head_disp_kernel(const _Float16* __restrict__ dp, float* __restrict__ outDisp,
                                 int t0, int tb) {
  int idx = blockIdx.x * 256 + threadIdx.x;
  if (idx >= tb * 4096) return;
  int tl = idx / 4096, r = idx % 4096;
  int y = r / 64, x = r % 64;
  float v = 0.0f;
  if (y >= 4 && y < 60 && x >= 4 && x < 60)
    v = (float)dp[((size_t)tl * 5) * 3136 + (y - 4) * 56 + (x - 4)];
  outDisp[(size_t)(t0 + tl) * 4096 + r] = v;
}

__global__ void head_conf_kernel(const _Float16* __restrict__ dp, float* __restrict__ conf, int tb) {
  int idx = blockIdx.x * 256 + threadIdx.x;
  if (idx >= tb * 3136) return;
  int tl = idx / 3136, pix = idx % 3136;
  float a[4], mx = -1e30f;
#pragma unroll
  for (int s = 0; s < 4; ++s) {
    a[s] = (float)dp[((size_t)tl * 5 + 1 + s) * 3136 + pix];
    mx = fmaxf(mx, a[s]);
  }
  float sum = 0.0f;
#pragma unroll
  for (int s = 0; s < 4; ++s) { a[s] = __expf(a[s] - mx); sum += a[s]; }
  float inv = 1.0f / sum;
#pragma unroll
  for (int s = 0; s < 4; ++s) conf[((size_t)tl * 4 + s) * 3136 + pix] = a[s] * inv;
}

// ------------------------------------------------------------------
// Confidence-weighted warp of the 4 source views with the predicted
// per-pixel disparity -> inter_lf [49][56][56].
// ------------------------------------------------------------------
__global__ void final_warp_kernel(const _Float16* __restrict__ dp, const float* __restrict__ conf,
                                  const float* __restrict__ img, const int* __restrict__ ind,
                                  float* __restrict__ outLF, int t0, int tb) {
  int idx = blockIdx.x * 256 + threadIdx.x;
  if (idx >= tb * 3136) return;
  int tl = idx / 3136, pix = idx % 3136;
  int t = t0 + tl;
  float d = (float)dp[((size_t)tl * 5) * 3136 + pix];
  int y = pix / 56, x = pix % 56;
  float acc = 0.0f;
#pragma unroll
  for (int s = 0; s < 4; ++s) {
    int iv = ind[s];
    float dy = (float)(iv / 7) - (float)(t / 7);
    float dx = (float)(iv % 7) - (float)(t % 7);
    float v = bilin64(img + s * 4096, (float)(x + 4) + d * dx, (float)(y + 4) + d * dy);
    acc += v * conf[((size_t)tl * 4 + s) * 3136 + pix];
  }
  outLF[(size_t)t * 3136 + pix] = acc;
}

// ------------------------------------------------------------------
// Host side. Input order (setup_inputs dict order, params flattened
// pp -> cp -> dp, each layer (w, b)):
//   0 ind_source, 1 img_source, 2..5 scalars, 6.. weight/bias pairs.
// ------------------------------------------------------------------
extern "C" void kernel_launch(void* const* d_in, const int* in_sizes, int n_in,
                              void* d_out, int out_size, void* d_ws, size_t ws_size,
                              hipStream_t stream) {
  (void)in_sizes; (void)n_in; (void)out_size;
  const int*   ind = (const int*)d_in[0];
  const float* img = (const float*)d_in[1];

  struct LD { int ic, oc, ks; };
  const LD L[11] = {{4,32,5},{32,64,5},{64,32,5},{32,4,5},
                    {64,200,3},{200,200,3},{200,16,5},
                    {16,64,5},{64,32,5},{32,16,3},{16,5,3}};

  size_t wOff[11], off = 0;
  for (int i = 0; i < 11; ++i) {
    wOff[i] = off;
    size_t bytes = (size_t)rup(L[i].oc, 16) * (size_t)rup(L[i].ic * L[i].ks * L[i].ks, 64) * 2;
    off = (off + bytes + 255) & ~(size_t)255;
  }
  const size_t wTot = off;

  // Pick the largest T-chunk that fits the workspace (deterministic).
  auto need = [&](int tb) -> size_t {
    return wTot + 2 * ((size_t)tb * 16 * 64 * 3136 * 2) + (size_t)tb * 4 * 3136 * 4 + 1024;
  };
  int TB = 1;
  if (ws_size >= need(49)) TB = 49;
  else if (ws_size >= need(7)) TB = 7;

  const size_t bufBytes = (size_t)TB * 16 * 64 * 3136 * 2;
  char* ws = (char*)d_ws;
  _Float16* actA = (_Float16*)(ws + wTot);
  _Float16* actB = (_Float16*)(ws + wTot + bufBytes);
  float*    conf = (float*)(ws + wTot + 2 * bufBytes);
  float* outDisp = (float*)d_out;
  float* outLF   = (float*)d_out + 49 * 4096;

  const float* Wp[11]; const float* Bp[11];
  for (int i = 0; i < 11; ++i) { Wp[i] = (const float*)d_in[6 + 2 * i]; Bp[i] = (const float*)d_in[7 + 2 * i]; }

#define PREP(i, ICv, OCv, KSv) do {                                              \
    int tot = rup(OCv, 16) * rup(ICv * KSv * KSv, 64);                           \
    prep_w_kernel<ICv, OCv, KSv><<<(tot + 255) / 256, 256, 0, stream>>>(         \
        Wp[i], (_Float16*)(ws + wOff[i])); } while (0)
  PREP(0, 4, 32, 5);  PREP(1, 32, 64, 5);  PREP(2, 64, 32, 5);  PREP(3, 32, 4, 5);
  PREP(4, 64, 200, 3); PREP(5, 200, 200, 3); PREP(6, 200, 16, 5);
  PREP(7, 16, 64, 5); PREP(8, 64, 32, 5);  PREP(9, 32, 16, 3);  PREP(10, 16, 5, 3);
#undef PREP

#define CONV(i, ICv, OCv, KSv, RELUv, INP, OUTP, Bn)                             \
    conv_wmma_kernel<ICv, OCv, KSv, RELUv><<<dim3(25, Bn), 256, 0, stream>>>(    \
        INP, (const _Float16*)(ws + wOff[i]), Bp[i], OUTP, Bn)

  for (int t0 = 0; t0 < 49; t0 += TB) {
    const int bpp = TB * 16;   // pp batch: (t, d) images
    const int bcd = TB;        // cp/dp batch: t images
    {
      int tot = TB * 16 * 4 * 3136;
      build_psv_kernel<<<(tot + 255) / 256, 256, 0, stream>>>(img, ind, actA, t0, TB);
    }
    CONV(0, 4, 32, 5, true,  actA, actB, bpp);
    CONV(1, 32, 64, 5, true, actB, actA, bpp);
    CONV(2, 64, 32, 5, true, actA, actB, bpp);
    CONV(3, 32, 4, 5, true,  actB, actA, bpp);   // [t][d*4+c] contiguity == cp reshape
    CONV(4, 64, 200, 3, true,  actA, actB, bcd);
    CONV(5, 200, 200, 3, true, actB, actA, bcd);
    CONV(6, 200, 16, 5, true,  actA, actB, bcd);
    CONV(7, 16, 64, 5, true,  actB, actA, bcd);
    CONV(8, 64, 32, 5, true,  actA, actB, bcd);
    CONV(9, 32, 16, 3, true,  actB, actA, bcd);
    CONV(10, 16, 5, 3, false, actA, actB, bcd);  // last dp layer: no ReLU
    {
      int tot = TB * 4096;
      head_disp_kernel<<<(tot + 255) / 256, 256, 0, stream>>>(actB, outDisp, t0, TB);
    }
    {
      int tot = TB * 3136;
      head_conf_kernel<<<(tot + 255) / 256, 256, 0, stream>>>(actB, conf, TB);
      final_warp_kernel<<<(tot + 255) / 256, 256, 0, stream>>>(actB, conf, img, ind, outLF, t0, TB);
    }
  }
#undef CONV
}
